// LatentRNNAgent_44298292691511
// MI455X (gfx1250) — matrix operand order repaired and checked by reading, not compiled, not run on hardware
//
#include <hip/hip_runtime.h>
#include <hip/hip_bf16.h>

// ---------------------------------------------------------------------------
// LatentRNNAgent fused hypernetwork on MI455X (gfx1250, wave32, WMMA bf16)
//   N=4096 samples, I=192, H=256, L=32, A=32
// All GEMMs mapped to V_WMMA_F32_16X16X32_BF16 (f32 accumulate).
// Round 2: WMMA C = inline zero (no splat movs); bias added post-WMMA so the
// ReLU lowers to a single v_max_num_f32 (canonical fadd input).
// ---------------------------------------------------------------------------

typedef __attribute__((ext_vector_type(16))) __bf16 v16bf;
typedef __attribute__((ext_vector_type(8)))  __bf16 v8bf;
typedef __attribute__((ext_vector_type(8)))  float  v8f;

#define N_SAMP 4096
#define I_DIM  192
#define H_DIM  256
#define L_DIM  32
#define A_DIM  32

static __device__ __forceinline__ v8f wmma_bf16(v16bf a, v16bf b, v8f c) {
  // 8 args: (neg_a, A, neg_b, B, c_mod, C, reuse_a, reuse_b)
  return __builtin_amdgcn_wmma_f32_16x16x32_bf16(false, a, false, b, (short)0, c,
                                                 false, false);
}

static __device__ __forceinline__ v8f v8f_zero() {
  v8f z = {};
  return z;
}

// A-matrix (16x32 bf16) tile of `latent` for samples [n0, n0+16).
// Layout: lane<16 -> row=lane, K = {0..15};
//         lane>=16 -> row=lane-16, K = {16..31}.
static __device__ __forceinline__ v16bf load_a_latent(const float* __restrict__ latent,
                                                      int n0, int lane) {
  const int row = lane & 15;
  const int kb  = (lane & 16) ? 16 : 0;
  v16bf a;
#pragma unroll
  for (int e = 0; e < 16; ++e) {
    const int k = kb + e;
    a[e] = (__bf16)latent[(size_t)(n0 + row) * L_DIM + k];
  }
  return a;
}

// Build 16x32 bf16 A tile from a row-major bf16 activation matrix [N][256]
// for K-chunk kc: two contiguous 16B loads per lane.
static __device__ __forceinline__ v16bf load_a_act(const __bf16* __restrict__ act,
                                                   int n0, int lane, int kc) {
  const int row = lane & 15;
  const int kb  = (lane & 16) ? 16 : 0;
  const __bf16* p = act + (size_t)(n0 + row) * H_DIM + kc * 32 + kb;
  v8bf lo = *(const v8bf*)p;
  v8bf hi = *(const v8bf*)(p + 8);
  return __builtin_shufflevector(lo, hi, 0, 1, 2, 3, 4, 5, 6, 7,
                                 8, 9, 10, 11, 12, 13, 14, 15);
}

// ---------------------------------------------------------------------------
// Packing kernels: pre-swizzle B operands into this file's WMMA B lane layout (contiguous k, as A).
// Tile = 32x16 bf16 (K x N), stored as 32 lanes x 16 bf16 (32B contiguous/lane).
//   lane<16:  b[e] = W[K=e,      N=lane]      lane>=16: b[e] = W[K=16+e, N=lane-16]
// Tile index t = coltile * kchunks + kc;  dst stride 512 bf16 per tile.
// ---------------------------------------------------------------------------
__global__ void pack_direct(const float* __restrict__ src, int ncols, int kchunks,
                            int ntiles, __bf16* __restrict__ dst) {
  const int idx = blockIdx.x * blockDim.x + threadIdx.x;
  if (idx >= ntiles * 32) return;
  const int lane = idx & 31;
  const int t    = idx >> 5;
  const int c    = t / kchunks;
  const int kc   = t - c * kchunks;
  const int kb   = kc * 32 + ((lane & 16) ? 16 : 0);
  const int col  = c * 16 + (lane & 15);
  v16bf o;
#pragma unroll
  for (int e = 0; e < 16; ++e) o[e] = (__bf16)src[(size_t)(kb + e) * ncols + col];
  *(v16bf*)(dst + (size_t)t * 512 + lane * 16) = o;
}

// Same, but B = src^T (src row-major [cols_total][ktot]), e.g. w_ih [768,256].
__global__ void pack_fromT(const float* __restrict__ src, int ktot, int kchunks,
                           int ntiles, __bf16* __restrict__ dst) {
  const int idx = blockIdx.x * blockDim.x + threadIdx.x;
  if (idx >= ntiles * 32) return;
  const int lane = idx & 31;
  const int t    = idx >> 5;
  const int c    = t / kchunks;
  const int kc   = t - c * kchunks;
  const int kb   = kc * 32 + ((lane & 16) ? 16 : 0);
  const int rowg = c * 16 + (lane & 15);
  v16bf o;
#pragma unroll
  for (int e = 0; e < 16; ++e) o[e] = (__bf16)src[(size_t)rowg * ktot + kb + e];
  *(v16bf*)(dst + (size_t)t * 512 + lane * 16) = o;
}

__global__ void f32_to_bf16(const float* __restrict__ src, __bf16* __restrict__ dst,
                            int n) {
  const int i = blockIdx.x * blockDim.x + threadIdx.x;
  if (i < n) dst[i] = (__bf16)src[i];
}

// ---------------------------------------------------------------------------
// Kernel 1: x = relu( einsum(inp, relu(latent@W1w+b1w)) + relu(latent@W1b+b1b) )
// Block = 16 waves (512 thr) = one 16-sample tile; wave w owns h-cols [16w,16w+16).
// 192 WMMAs per wave (one per i); C = inline 0, bias added post-WMMA.
// ---------------------------------------------------------------------------
__launch_bounds__(512)
__global__ void layer1_kernel(const float* __restrict__ inputs,
                              const float* __restrict__ latent,
                              const float* __restrict__ b1w,
                              const float* __restrict__ b1b,
                              const __bf16* __restrict__ pW1w,
                              const __bf16* __restrict__ pW1b,
                              __bf16* __restrict__ xbf) {
  __shared__ float inpLDS[I_DIM * 16];  // [i][m], 12 KB
  const int n0   = blockIdx.x * 16;
  const int tid  = threadIdx.x;
  const int wave = tid >> 5;
  const int lane = tid & 31;

  for (int idx = tid; idx < I_DIM * 16; idx += 512) {
    const int m = idx & 15, i = idx >> 4;
    inpLDS[idx] = inputs[(size_t)(n0 + m) * I_DIM + i];
  }
  __syncthreads();

  const int col = lane & 15;
  const int mB  = (lane & 16) ? 8 : 0;  // C/D row base for this lane half
  const v16bf a = load_a_latent(latent, n0, lane);

  const int ht = wave;
  v8f acc = v8f_zero();
  const v8f zc = v8f_zero();  // selects inline SRC2=0 in v_wmma

  for (int i = 0; i < I_DIM; ++i) {
    const v16bf b =
        *(const v16bf*)(pW1w + (size_t)(i * 16 + ht) * 512 + lane * 16);
    const float bias = b1w[i * H_DIM + ht * 16 + col];
    const v8f t = wmma_bf16(a, b, zc);
    const float4 sl = *(const float4*)&inpLDS[i * 16 + mB];
    const float4 sh = *(const float4*)&inpLDS[i * 16 + mB + 4];
    const float s[8] = {sl.x, sl.y, sl.z, sl.w, sh.x, sh.y, sh.z, sh.w};
#pragma unroll
    for (int v = 0; v < 8; ++v)
      acc[v] += s[v] * fmaxf(t[v] + bias, 0.f);  // fadd -> single v_max_num
  }

  // fc1_b = relu(latent@W1b + b1b), then x = relu(acc + fc1_b)
  const v16bf bb = *(const v16bf*)(pW1b + (size_t)ht * 512 + lane * 16);
  const float bias2 = b1b[ht * 16 + col];
  const v8f t2 = wmma_bf16(a, bb, zc);
#pragma unroll
  for (int v = 0; v < 8; ++v) {
    const float x = fmaxf(acc[v] + fmaxf(t2[v] + bias2, 0.f), 0.f);
    xbf[(size_t)(n0 + mB + v) * H_DIM + ht * 16 + col] = (__bf16)x;
  }
}

// ---------------------------------------------------------------------------
// Kernel 2: GRU cell. Wave w owns h-cols [16w,16w+16) -> r,z,n tiles land in
// the same wave (col-tiles w, 16+w, 32+w of the [16,768] GEMMs). 48 WMMAs/wave,
// accumulators chained through the WMMA C operand (no extra VALU).
// ---------------------------------------------------------------------------
__launch_bounds__(512)
__global__ void gru_kernel(const float* __restrict__ hidden,
                           const float* __restrict__ b_ih,
                           const float* __restrict__ b_hh,
                           const __bf16* __restrict__ xbf,
                           const __bf16* __restrict__ hidbf,
                           const __bf16* __restrict__ pWih,
                           const __bf16* __restrict__ pWhh,
                           float* __restrict__ hout) {
  const int n0   = blockIdx.x * 16;
  const int tid  = threadIdx.x;
  const int w    = tid >> 5;
  const int lane = tid & 31;
  const int col  = lane & 15;
  const int mB   = (lane & 16) ? 8 : 0;

  v8f gi[3], gh[3];
#pragma unroll
  for (int g = 0; g < 3; ++g) { gi[g] = v8f_zero(); gh[g] = v8f_zero(); }

  for (int kc = 0; kc < 8; ++kc) {
    const v16bf ax = load_a_act(xbf,   n0, lane, kc);
    const v16bf ah = load_a_act(hidbf, n0, lane, kc);
#pragma unroll
    for (int g = 0; g < 3; ++g) {
      const int t = ((g * 16 + w) * 8 + kc);
      const v16bf bih = *(const v16bf*)(pWih + (size_t)t * 512 + lane * 16);
      gi[g] = wmma_bf16(ax, bih, gi[g]);
      const v16bf bhh = *(const v16bf*)(pWhh + (size_t)t * 512 + lane * 16);
      gh[g] = wmma_bf16(ah, bhh, gh[g]);
    }
  }

  const int hcol = w * 16 + col;
  const float bir = b_ih[hcol],             bhr = b_hh[hcol];
  const float biz = b_ih[H_DIM + hcol],     bhz = b_hh[H_DIM + hcol];
  const float bin = b_ih[2 * H_DIM + hcol], bhn = b_hh[2 * H_DIM + hcol];

#pragma unroll
  for (int v = 0; v < 8; ++v) {
    const int m = mB + v;
    const float rr = (gi[0][v] + bir) + (gh[0][v] + bhr);
    const float zz = (gi[1][v] + biz) + (gh[1][v] + bhz);
    const float r  = 1.f / (1.f + __expf(-rr));
    const float z  = 1.f / (1.f + __expf(-zz));
    const float nn = tanhf((gi[2][v] + bin) + r * (gh[2][v] + bhn));
    const float hp = hidden[(size_t)(n0 + m) * H_DIM + hcol];
    hout[(size_t)(n0 + m) * H_DIM + hcol] = (1.f - z) * nn + z * hp;
  }
}

// ---------------------------------------------------------------------------
// Kernel 3: q = relu( einsum(h_new, relu(latent@W2w+b2w)) + relu(latent@W2b+b2b) )
// Wave w covers h in [16w,16w+16); partial q tiles reduced via LDS ds_add_f32.
// ---------------------------------------------------------------------------
__launch_bounds__(512)
__global__ void layer2_kernel(const float* __restrict__ latent,
                              const float* __restrict__ b2w,
                              const float* __restrict__ b2b,
                              const __bf16* __restrict__ pW2w,
                              const __bf16* __restrict__ pW2b,
                              const float* __restrict__ hout,
                              float* __restrict__ qout) {
  __shared__ float hLDS[H_DIM * 16];   // [h][m], 16 KB
  __shared__ float qLDS[16 * A_DIM];   // [m][a], 2 KB
  const int n0   = blockIdx.x * 16;
  const int tid  = threadIdx.x;
  const int w    = tid >> 5;
  const int lane = tid & 31;
  const int col  = lane & 15;
  const int mB   = (lane & 16) ? 8 : 0;

  for (int idx = tid; idx < H_DIM * 16; idx += 512) {
    const int m = idx & 15, h = idx >> 4;
    hLDS[idx] = hout[(size_t)(n0 + m) * H_DIM + h];
  }
  qLDS[tid] = 0.f;  // 512 == 16*32
  __syncthreads();

  const v16bf a = load_a_latent(latent, n0, lane);
  const v8f zc = v8f_zero();
  v8f acc[2] = {v8f_zero(), v8f_zero()};

  for (int j = 0; j < 16; ++j) {
    const int h = w * 16 + j;
    const float4 sl = *(const float4*)&hLDS[h * 16 + mB];
    const float4 sh = *(const float4*)&hLDS[h * 16 + mB + 4];
    const float s[8] = {sl.x, sl.y, sl.z, sl.w, sh.x, sh.y, sh.z, sh.w};
#pragma unroll
    for (int at = 0; at < 2; ++at) {
      const v16bf b =
          *(const v16bf*)(pW2w + (size_t)(h * 2 + at) * 512 + lane * 16);
      const float bias = b2w[h * A_DIM + at * 16 + col];
      const v8f t = wmma_bf16(a, b, zc);
#pragma unroll
      for (int v = 0; v < 8; ++v)
        acc[at][v] += s[v] * fmaxf(t[v] + bias, 0.f);
    }
  }

#pragma unroll
  for (int at = 0; at < 2; ++at)
#pragma unroll
    for (int v = 0; v < 8; ++v)
      atomicAdd(&qLDS[(mB + v) * A_DIM + at * 16 + col], acc[at][v]);
  __syncthreads();

  if (w < 2) {  // wave-uniform branch: EXEC stays all-ones for the WMMA
    const v16bf b = *(const v16bf*)(pW2b + (size_t)w * 512 + lane * 16);
    const float bias = b2b[w * 16 + col];
    const v8f t2 = wmma_bf16(a, b, zc);
#pragma unroll
    for (int v = 0; v < 8; ++v) {
      const int m = mB + v;
      const float q =
          fmaxf(qLDS[m * A_DIM + w * 16 + col] + fmaxf(t2[v] + bias, 0.f), 0.f);
      qout[(size_t)(n0 + m) * A_DIM + w * 16 + col] = q;
    }
  }
}

// ---------------------------------------------------------------------------
extern "C" void kernel_launch(void* const* d_in, const int* in_sizes, int n_in,
                              void* d_out, int out_size, void* d_ws, size_t ws_size,
                              hipStream_t stream) {
  const float* inputs = (const float*)d_in[0];
  const float* hidden = (const float*)d_in[1];
  const float* latent = (const float*)d_in[2];
  const float* W1w    = (const float*)d_in[3];
  const float* b1w    = (const float*)d_in[4];
  const float* W1b    = (const float*)d_in[5];
  const float* b1b    = (const float*)d_in[6];
  const float* W2w    = (const float*)d_in[7];
  const float* b2w    = (const float*)d_in[8];
  const float* W2b    = (const float*)d_in[9];
  const float* b2b    = (const float*)d_in[10];
  const float* w_ih   = (const float*)d_in[11];
  const float* w_hh   = (const float*)d_in[12];
  const float* b_ih   = (const float*)d_in[13];
  const float* b_hh   = (const float*)d_in[14];

  float* qout = (float*)d_out;                              // [N, A]
  float* hout = (float*)d_out + (size_t)N_SAMP * A_DIM;     // [N, H]

  char* ws = (char*)d_ws;
  __bf16* pW1w  = (__bf16*)(ws + 0);        // 3072 tiles * 1 KB = 3 MB
  __bf16* pW1b  = (__bf16*)(ws + 3145728);  // 16 KB
  __bf16* pWih  = (__bf16*)(ws + 3162112);  // 384 KB
  __bf16* pWhh  = (__bf16*)(ws + 3555328);  // 384 KB
  __bf16* pW2w  = (__bf16*)(ws + 3948544);  // 512 KB
  __bf16* pW2b  = (__bf16*)(ws + 4472832);  // 2 KB
  __bf16* xbf   = (__bf16*)(ws + 4474880);  // 2 MB
  __bf16* hidbf = (__bf16*)(ws + 6572032);  // 2 MB  (total ~8.7 MB)

  // --- pack B operands into this file's WMMA B lane layout (contiguous k) (runs every call; cheap) ---
  pack_direct<<<(3072 * 32 + 255) / 256, 256, 0, stream>>>(W1w, I_DIM * H_DIM, 1,
                                                           3072, pW1w);
  pack_direct<<<(16 * 32 + 255) / 256, 256, 0, stream>>>(W1b, H_DIM, 1, 16, pW1b);
  pack_direct<<<(512 * 32 + 255) / 256, 256, 0, stream>>>(W2w, H_DIM * A_DIM, 1,
                                                          512, pW2w);
  pack_direct<<<1, 64, 0, stream>>>(W2b, A_DIM, 1, 2, pW2b);
  pack_fromT<<<(384 * 32 + 255) / 256, 256, 0, stream>>>(w_ih, H_DIM, 8, 384, pWih);
  pack_fromT<<<(384 * 32 + 255) / 256, 256, 0, stream>>>(w_hh, H_DIM, 8, 384, pWhh);
  f32_to_bf16<<<(N_SAMP * H_DIM + 255) / 256, 256, 0, stream>>>(
      hidden, hidbf, N_SAMP * H_DIM);

  // --- fused pipeline ---
  layer1_kernel<<<N_SAMP / 16, 512, 0, stream>>>(inputs, latent, b1w, b1b, pW1w,
                                                 pW1b, xbf);
  gru_kernel<<<N_SAMP / 16, 512, 0, stream>>>(hidden, b_ih, b_hh, xbf, hidbf,
                                              pWih, pWhh, hout);
  layer2_kernel<<<N_SAMP / 16, 512, 0, stream>>>(latent, b2w, b2b, pW2w, pW2b,
                                                 hout, qout);
}
